// PerAssetSpikingEncoder_12438225289663
// MI455X (gfx1250) — compile-verified, matrix operand order checked by reading, not run on hardware
//
#include <hip/hip_runtime.h>
#include <hip/hip_bf16.h>

typedef __attribute__((ext_vector_type(16))) __bf16 v16bf;
typedef __attribute__((ext_vector_type(8)))  float  v8f;

#define NB    256
#define NT    2048
#define DIN   64
#define DH    128
#define CHUNK 16

// sigmoid(5*(m-1)) = rcp(1 + exp2(-5*log2(e)*(m-1)))
#define NEG5_LOG2E (-7.2134752044f)

__device__ __forceinline__ void split_bf16(float v, __bf16& hi, __bf16& lo) {
    hi = (__bf16)v;
    lo = (__bf16)(v - (float)hi);
}

__global__ __launch_bounds__(128)
void spiking_encoder_fused(const float* __restrict__ x,    // [B,T,DIN]
                           const float* __restrict__ W,    // [DH,DIN]
                           const float* __restrict__ bias, // [DH]
                           float* __restrict__ out)        // [B,T,DH]
{
    // One batch element per block, 128 threads = 4 waves.
    __shared__ float xs [CHUNK * DIN];  // 4 KB: staged x tile (f32)
    __shared__ float ffs[CHUNK * DH];   // 8 KB: ff tile (f32)

    const int b    = blockIdx.x;
    const int tid  = threadIdx.x;       // 0..127
    const int lane = tid & 31;          // wave32
    const int wv   = tid >> 5;          // 4 waves; wave owns h in [wv*32, wv*32+32)
    const int n    = lane & 15;         // tile row/col index
    const int kh   = lane >> 4;         // K-half select within lane group

    // ---- B-matrix (W^T) fragments, hi/lo bf16 split, chunk-invariant ----
    // 16-bit 32x16 layout: lane covers n = lane%16; VGPR r holds K pair at
    // k = 16*(r/4) + 8*(lane/16) + 2*(r%4). j = h-tile {0,1}, s = K-half {0,1}.
    const int h0 = wv * 32 + n;
    v16bf whi[2][2], wlo[2][2];
#pragma unroll
    for (int j = 0; j < 2; ++j) {
        const int h = h0 + 16 * j;
#pragma unroll
        for (int s = 0; s < 2; ++s) {
#pragma unroll
            for (int r = 0; r < 8; ++r) {
                const int k = 32 * s + ((r >> 2) << 4) + (kh << 3) + ((r & 3) << 1);
                __bf16 hi0, lo0, hi1, lo1;
                split_bf16(W[h * DIN + k],     hi0, lo0);
                split_bf16(W[h * DIN + k + 1], hi1, lo1);
                whi[j][s][2*r] = hi0;  whi[j][s][2*r+1] = hi1;
                wlo[j][s][2*r] = lo0;  wlo[j][s][2*r+1] = lo1;
            }
        }
    }

    // Recurrence state: all 128 threads, h = tid.
    float mem = 0.0f, spk = 0.0f;
    const float bia = bias[tid];

    const float* xb = x   + (size_t)b * NT * DIN;
    float*       ob = out + (size_t)b * NT * DH;

    // Software-pipelined x staging: prefetch chunk 0 into registers.
    float4 pf0 = ((const float4*)xb)[tid];
    float4 pf1 = ((const float4*)xb)[tid + 128];

    for (int t0 = 0; t0 < NT; t0 += CHUNK) {
        // ---- deposit current tile; immediately prefetch next chunk ----
        ((float4*)xs)[tid]       = pf0;
        ((float4*)xs)[tid + 128] = pf1;
        if (t0 + CHUNK < NT) {
            const float4* nxt = (const float4*)(xb + (size_t)(t0 + CHUNK) * DIN);
            pf0 = nxt[tid];          // latency overlaps WMMA + scan below
            pf1 = nxt[tid + 128];
        }
        __syncthreads();

        // ---- ff(16x128) = xtile(16x64) x W^T(64x128), hi/lo split WMMA ----
        {
            v16bf ahi[2], alo[2];
            const int m = n;  // A row = lane%16
#pragma unroll
            for (int s = 0; s < 2; ++s) {
#pragma unroll
                for (int r = 0; r < 8; ++r) {
                    const int k = 32 * s + ((r >> 2) << 4) + (kh << 3) + ((r & 3) << 1);
                    __bf16 hi0, lo0, hi1, lo1;
                    split_bf16(xs[m * DIN + k],     hi0, lo0);
                    split_bf16(xs[m * DIN + k + 1], hi1, lo1);
                    ahi[s][2*r] = hi0;  ahi[s][2*r+1] = hi1;
                    alo[s][2*r] = lo0;  alo[s][2*r+1] = lo1;
                }
            }
#pragma unroll
            for (int j = 0; j < 2; ++j) {
                v8f c = {};
#pragma unroll
                for (int s = 0; s < 2; ++s) {
                    // hi*hi + lo*hi + hi*lo  (lo*lo negligible)
                    c = __builtin_amdgcn_wmma_f32_16x16x32_bf16(
                            false, ahi[s], false, whi[j][s], (short)0, c, false, false);
                    c = __builtin_amdgcn_wmma_f32_16x16x32_bf16(
                            false, alo[s], false, whi[j][s], (short)0, c, false, false);
                    c = __builtin_amdgcn_wmma_f32_16x16x32_bf16(
                            false, ahi[s], false, wlo[j][s], (short)0, c, false, false);
                }
                // C layout: VGPR r -> row m = r + 8*(lane/16), col n = lane%16
#pragma unroll
                for (int r = 0; r < 8; ++r) {
                    ffs[(r + 8 * kh) * DH + h0 + 16 * j] = c[r];
                }
            }
        }
        __syncthreads();

        // ---- sequential spiking recurrence: 16 steps, all threads active ----
#pragma unroll
        for (int t = 0; t < CHUNK; ++t) {
            const float ff = ffs[t * DH + tid] + bia;
            mem = 0.9f * mem + ff - spk;
            const float e = __builtin_amdgcn_exp2f(NEG5_LOG2E * (mem - 1.0f));
            spk = __builtin_amdgcn_rcpf(1.0f + e);       // v_exp + v_rcp only
            ob[(size_t)(t0 + t) * DH + tid] = spk;       // 512B coalesced per t
        }
        __syncthreads();  // protect xs/ffs before next chunk overwrites them
    }
}

extern "C" void kernel_launch(void* const* d_in, const int* in_sizes, int n_in,
                              void* d_out, int out_size, void* d_ws, size_t ws_size,
                              hipStream_t stream) {
    const float* x    = (const float*)d_in[0];  // [256, 2048, 64]
    const float* W    = (const float*)d_in[1];  // [128, 64]
    const float* bias = (const float*)d_in[2];  // [128]
    float*       out  = (float*)d_out;          // [256, 2048, 128]
    (void)in_sizes; (void)n_in; (void)out_size; (void)d_ws; (void)ws_size;

    spiking_encoder_fused<<<NB, 128, 0, stream>>>(x, W, bias, out);
}
